// RecurrentBeamSNN_1451698946518
// MI455X (gfx1250) — matrix operand order repaired and checked by reading, not compiled, not run on hardware
//
#include <hip/hip_runtime.h>
#include <hip/hip_bf16.h>

// RecurrentBeamSNN on gfx1250 (MI455X):
//  - 32 workgroups x 256 threads (8 wave32); block b handles batch rows [16b,16b+16)
//  - weights pre-swizzled to WMMA B-fragment layout in LDS (f16), ~170KB LDS/WGP
//  - recurrent state (mem1/mem2/mem_out) lives in registers across all T=512 steps
//  - x pre-converted to f16 in d_ws; per-step tiles double-buffered into LDS via
//    GLOBAL_LOAD_ASYNC_TO_LDS_B128 (ASYNCcnt), overlapped with WMMA compute
//  - all matmuls via v_wmma_f32_16x16x32_f16

typedef __attribute__((ext_vector_type(16))) _Float16 v16h;
typedef __attribute__((ext_vector_type(8)))  _Float16 v8h;
typedef __attribute__((ext_vector_type(8)))  float    v8f;

#define B_  512
#define T_  512
#define F_  128
#define H1_ 256
#define H2_ 128
#define NB_ 64

union Frag { v16h v; v8h h[2]; };

__device__ __forceinline__ v8f wmma16(v16h a, v16h b, v8f c) {
  return __builtin_amdgcn_wmma_f32_16x16x32_f16(
      /*neg_a=*/false, a, /*neg_b=*/false, b,
      /*c_mod=*/(short)0, c, /*reuse_a=*/false, /*reuse_b=*/false);
}

// B fragment: 16 contiguous halves per lane (pre-swizzled in LDS) -> 2x ds_load_b128
__device__ __forceinline__ v16h ldsB(const _Float16* p) {
  Frag f;
  f.h[0] = *(const v8h*)(p);
  f.h[1] = *(const v8h*)(p + 8);
  return f.v;
}

// A fragment from row-major [M][K] f16 LDS buffer.
// Lane (0..31): M = lane&15, kb8 = (lane>>4)*8.
// halves 0..15 <- K = k0 + 2*kb8 + 0..15
__device__ __forceinline__ v16h ldsA_rm(const _Float16* row, int k0, int kb8) {
  Frag f;
  f.h[0] = *(const v8h*)(row + k0 + 2 * kb8);
  f.h[1] = *(const v8h*)(row + k0 + 2 * kb8 + 8);
  return f.v;
}

// Low 32 bits of a generic (shared-aperture) address = LDS byte offset.
__device__ __forceinline__ uint32_t lds_addr(const void* p) {
  return (uint32_t)(uintptr_t)p;
}

// Async DMA: 16 bytes global -> LDS, tracked by ASYNCcnt (GVS mode: sgpr base + vgpr offset).
__device__ __forceinline__ void async_ld_b128(uint32_t lds_off, uint32_t goff,
                                              const _Float16* base) {
  asm volatile("global_load_async_to_lds_b128 %0, %1, %2"
               :: "v"(lds_off), "v"(goff), "s"(base)
               : "memory");
}
__device__ __forceinline__ void wait_async0() {
  asm volatile("s_wait_asynccnt 0x0" ::: "memory");
}

// ---------------- pre-pass: x f32 -> f16 into workspace ----------------
__global__ __launch_bounds__(256) void cvt_x_kernel(const float* __restrict__ x,
                                                    _Float16* __restrict__ xh) {
  int i = blockIdx.x * 256 + threadIdx.x;       // each thread converts 8 elements
  const float4* p = (const float4*)x + (size_t)i * 2;
  float4 a = p[0], b = p[1];
  v8h h;
  h[0]=(_Float16)a.x; h[1]=(_Float16)a.y; h[2]=(_Float16)a.z; h[3]=(_Float16)a.w;
  h[4]=(_Float16)b.x; h[5]=(_Float16)b.y; h[6]=(_Float16)b.z; h[7]=(_Float16)b.w;
  *((v8h*)xh + i) = h;
}

__global__ __launch_bounds__(256) void snn_kernel(
    const float* __restrict__ x,
    const float* __restrict__ W1, const float* __restrict__ b1,
    const float* __restrict__ g1, const float* __restrict__ bt1,
    const float* __restrict__ W2, const float* __restrict__ b2,
    const float* __restrict__ g2, const float* __restrict__ bt2,
    const float* __restrict__ Wo, const float* __restrict__ bo,
    const _Float16* __restrict__ xh, int use_async,
    float* __restrict__ out)
{
  // Weights packed as [kt][nt][lane][16 halves] (WMMA B-fragment order).
  __shared__ __align__(32) _Float16 sW1[4 * 16 * 32 * 16]; // 64 KB
  __shared__ __align__(32) _Float16 sW2[8 *  8 * 32 * 16]; // 64 KB
  __shared__ __align__(32) _Float16 sWo[4 *  4 * 32 * 16]; // 16 KB
  __shared__ __align__(32) _Float16 sX[2][16 * F_];        // x tile ping-pong, 8 KB
  __shared__ __align__(32) _Float16 sS1[16 * H1_];         // spk1, row-major
  __shared__ __align__(32) _Float16 sS2[16 * H2_];         // spk2, row-major
  __shared__ float sB1[H1_], sG1[H1_], sT1[H1_];
  __shared__ float sB2[H2_], sG2[H2_], sT2[H2_];
  __shared__ float sBo[NB_];
  __shared__ float sRed[2][8][16];   // [sum|sumsq][wave][row]
  __shared__ float sMean[16], sRstd[16];

  const int tid  = threadIdx.x;
  const int w    = tid >> 5;    // wave 0..7
  const int lane = tid & 31;
  const int nl   = lane & 15;   // N within tile / M for A frags
  const int hi   = lane >> 4;
  const int kb8  = hi * 8;
  const int rowb = hi * 8;      // M base for C-matrix registers
  const int b0   = blockIdx.x * 16;

  // staging geometry: thread -> (row, 8-half segment) of the 16x128 x tile
  const int srow = tid >> 4;
  const int sseg = (tid & 15) * 8;

  // ---- stage x(t=0) tile (overlaps with weight packing below) ----
  if (use_async) {
    uint32_t goff = (uint32_t)(((b0 + srow) * T_ + 0) * F_ + sseg) * 2u;
    async_ld_b128(lds_addr(&sX[0][srow * F_ + sseg]), goff, xh);
  } else {
    const float* src = x + ((size_t)(b0 + srow) * T_ + 0) * F_ + sseg;
    float4 f0 = *(const float4*)src, f1 = *(const float4*)(src + 4);
    v8h h;
    h[0]=(_Float16)f0.x; h[1]=(_Float16)f0.y; h[2]=(_Float16)f0.z; h[3]=(_Float16)f0.w;
    h[4]=(_Float16)f1.x; h[5]=(_Float16)f1.y; h[6]=(_Float16)f1.z; h[7]=(_Float16)f1.w;
    *(v8h*)&sX[0][srow * F_ + sseg] = h;
  }

  // ---------------- one-time: stage + swizzle weights into LDS ----------------
  for (int j = tid; j < F_ * H1_; j += 256) {          // W1 [128][256]
    int K = j >> 8, N = j & 255;
    int kt = K >> 5, kr = K & 31;
    int ln = (kr >> 4) * 16 + (N & 15);
    sW1[(((kt * 16 + (N >> 4)) * 32) + ln) * 16 + (kr & 15)] = (_Float16)W1[j];
  }
  for (int j = tid; j < H1_ * H2_; j += 256) {         // W2 [256][128]
    int K = j >> 7, N = j & 127;
    int kt = K >> 5, kr = K & 31;
    int ln = (kr >> 4) * 16 + (N & 15);
    sW2[(((kt * 8 + (N >> 4)) * 32) + ln) * 16 + (kr & 15)] = (_Float16)W2[j];
  }
  for (int j = tid; j < H2_ * NB_; j += 256) {         // Wout [128][64]
    int K = j >> 6, N = j & 63;
    int kt = K >> 5, kr = K & 31;
    int ln = (kr >> 4) * 16 + (N & 15);
    sWo[(((kt * 4 + (N >> 4)) * 32) + ln) * 16 + (kr & 15)] = (_Float16)Wo[j];
  }
  { sB1[tid] = b1[tid]; sG1[tid] = g1[tid]; sT1[tid] = bt1[tid]; }
  if (tid < H2_) { sB2[tid] = b2[tid]; sG2[tid] = g2[tid]; sT2[tid] = bt2[tid]; }
  if (tid < NB_) { sBo[tid] = bo[tid]; }
  if (use_async) wait_async0();
  __syncthreads();

  // ---------------- recurrent state in registers ----------------
  float m1a[8], m1b[8], m2[8], mo[8];
#pragma unroll
  for (int r = 0; r < 8; ++r) { m1a[r] = 0.f; m1b[r] = 0.f; m2[r] = 0.f; mo[r] = 0.f; }

  float* outLog = out;                                   // (B,T,NB)
  float* outSpk = out + (size_t)B_ * T_ * NB_;           // (B,T,H2)
  float* outM1  = outSpk + (size_t)B_ * T_ * H2_;        // (B,H1)
  float* outM2  = outM1 + (size_t)B_ * H1_;              // (B,H2)
  float* outMo  = outM2 + (size_t)B_ * H2_;              // (B,NB)

  float s[8], q[8];

  for (int t = 0; t < T_; ++t) {
    // ---- layer 1: (16x128)@(128x256); this wave -> H1 columns [32w, 32w+32) ----
    v8f c0 = {}; v8f c1 = {};
    {
      const _Float16* xrow = &sX[t & 1][nl * F_];
#pragma unroll
      for (int kt = 0; kt < 4; ++kt) {
        v16h a   = ldsA_rm(xrow, kt * 32, kb8);
        v16h bb0 = ldsB(&sW1[(((kt * 16 + (w * 2    )) * 32) + lane) * 16]);
        v16h bb1 = ldsB(&sW1[(((kt * 16 + (w * 2 + 1)) * 32) + lane) * 16]);
        c0 = wmma16(a, bb0, c0);
        c1 = wmma16(a, bb1, c1);
      }
    }

    // ---- kick off staging of x(t+1) into the other LDS buffer (overlaps LN/L2) ----
    // sX[(t+1)&1] was last read in iteration t-1's layer 1, so writing now is safe.
    if (t + 1 < T_) {
      const int nb = (t + 1) & 1;
      if (use_async) {
        uint32_t goff = (uint32_t)(((b0 + srow) * T_ + (t + 1)) * F_ + sseg) * 2u;
        async_ld_b128(lds_addr(&sX[nb][srow * F_ + sseg]), goff, xh);
      } else {
        const float* src = x + ((size_t)(b0 + srow) * T_ + (t + 1)) * F_ + sseg;
        float4 f0 = *(const float4*)src, f1 = *(const float4*)(src + 4);
        v8h h;
        h[0]=(_Float16)f0.x; h[1]=(_Float16)f0.y; h[2]=(_Float16)f0.z; h[3]=(_Float16)f0.w;
        h[4]=(_Float16)f1.x; h[5]=(_Float16)f1.y; h[6]=(_Float16)f1.z; h[7]=(_Float16)f1.w;
        *(v8h*)&sX[nb][srow * F_ + sseg] = h;
      }
    }

    {
      float bias0 = sB1[w * 32 + nl], bias1 = sB1[w * 32 + 16 + nl];
#pragma unroll
      for (int r = 0; r < 8; ++r) {
        float v0 = c0[r] + bias0, v1 = c1[r] + bias1;
        c0[r] = v0; c1[r] = v1;
        s[r] = v0 + v1; q[r] = v0 * v0 + v1 * v1;
      }
    }
#pragma unroll
    for (int r = 0; r < 8; ++r) {
#pragma unroll
      for (int m = 1; m < 16; m <<= 1) {       // reduce within 16-lane half (rows fixed)
        s[r] += __shfl_xor(s[r], m, 32);
        q[r] += __shfl_xor(q[r], m, 32);
      }
    }
    if (nl == 0) {
#pragma unroll
      for (int r = 0; r < 8; ++r) { sRed[0][w][rowb + r] = s[r]; sRed[1][w][rowb + r] = q[r]; }
    }
    __syncthreads();
    if (tid < 16) {
      float ts = 0.f, tq = 0.f;
#pragma unroll
      for (int ww = 0; ww < 8; ++ww) { ts += sRed[0][ww][tid]; tq += sRed[1][ww][tid]; }
      float mean = ts * (1.0f / H1_);
      float var  = tq * (1.0f / H1_) - mean * mean;
      sMean[tid] = mean;
      sRstd[tid] = rsqrtf(var + 1e-5f);
    }
    __syncthreads();
    {
      float ga = sG1[w * 32 + nl]      * 5.0f, ta = sT1[w * 32 + nl]      * 5.0f;
      float gb = sG1[w * 32 + 16 + nl] * 5.0f, tb = sT1[w * 32 + 16 + nl] * 5.0f;
#pragma unroll
      for (int r = 0; r < 8; ++r) {
        float mean = sMean[rowb + r], rstd = sRstd[rowb + r];
        float cur0 = (c0[r] - mean) * rstd * ga + ta;
        float cur1 = (c1[r] - mean) * rstd * gb + tb;
        float p0 = m1a[r]; float n0 = (p0 > 1.0f) ? 0.0f : (0.9f * p0 + cur0);
        float p1 = m1b[r]; float n1 = (p1 > 1.0f) ? 0.0f : (0.9f * p1 + cur1);
        m1a[r] = n0; m1b[r] = n1;
        sS1[(rowb + r) * H1_ + w * 32 + nl]      = (_Float16)((n0 > 1.0f) ? 1.0f : 0.0f);
        sS1[(rowb + r) * H1_ + w * 32 + 16 + nl] = (_Float16)((n1 > 1.0f) ? 1.0f : 0.0f);
      }
    }
    __syncthreads();

    // ---- layer 2: (16x256)@(256x128); this wave -> H2 columns [16w, 16w+16) ----
    v8f c2 = {};
    {
      const _Float16* a2row = sS1 + nl * H1_;
#pragma unroll
      for (int kt = 0; kt < 8; ++kt) {
        v16h a  = ldsA_rm(a2row, kt * 32, kb8);
        v16h bb = ldsB(&sW2[(((kt * 8 + w) * 32) + lane) * 16]);
        c2 = wmma16(a, bb, c2);
      }
    }
    {
      float bias2 = sB2[w * 16 + nl];
#pragma unroll
      for (int r = 0; r < 8; ++r) {
        float v = c2[r] + bias2; c2[r] = v;
        s[r] = v; q[r] = v * v;
      }
    }
#pragma unroll
    for (int r = 0; r < 8; ++r) {
#pragma unroll
      for (int m = 1; m < 16; m <<= 1) {
        s[r] += __shfl_xor(s[r], m, 32);
        q[r] += __shfl_xor(q[r], m, 32);
      }
    }
    if (nl == 0) {
#pragma unroll
      for (int r = 0; r < 8; ++r) { sRed[0][w][rowb + r] = s[r]; sRed[1][w][rowb + r] = q[r]; }
    }
    __syncthreads();
    if (tid < 16) {
      float ts = 0.f, tq = 0.f;
#pragma unroll
      for (int ww = 0; ww < 8; ++ww) { ts += sRed[0][ww][tid]; tq += sRed[1][ww][tid]; }
      float mean = ts * (1.0f / H2_);
      float var  = tq * (1.0f / H2_) - mean * mean;
      sMean[tid] = mean;
      sRstd[tid] = rsqrtf(var + 1e-5f);
    }
    __syncthreads();
    {
      float g = sG2[w * 16 + nl] * 5.0f, tb2 = sT2[w * 16 + nl] * 5.0f;
#pragma unroll
      for (int r = 0; r < 8; ++r) {
        float mean = sMean[rowb + r], rstd = sRstd[rowb + r];
        float cur = (c2[r] - mean) * rstd * g + tb2;
        float p = m2[r]; float n = (p > 1.0f) ? 0.0f : (0.9f * p + cur);
        m2[r] = n;
        float sp = (n > 1.0f) ? 1.0f : 0.0f;
        outSpk[((size_t)(b0 + rowb + r) * T_ + t) * H2_ + w * 16 + nl] = sp;
        sS2[(rowb + r) * H2_ + w * 16 + nl] = (_Float16)sp;
      }
    }
    // x(t+1) must be LDS-resident & visible to all waves past this barrier.
    if (use_async && (t + 1 < T_)) wait_async0();
    __syncthreads();

    // ---- output layer: (16x128)@(128x64); waves 0..3 -> NB columns [16w, 16w+16) ----
    // Waves 4..7 run ahead into next timestep's layer-1 WMMAs (overlap).
    if (w < 4) {
      v8f co = {};
      const _Float16* aorow = sS2 + nl * H2_;
#pragma unroll
      for (int kt = 0; kt < 4; ++kt) {
        v16h a  = ldsA_rm(aorow, kt * 32, kb8);
        v16h bb = ldsB(&sWo[(((kt * 4 + w) * 32) + lane) * 16]);
        co = wmma16(a, bb, co);
      }
      float biaso = sBo[w * 16 + nl];
#pragma unroll
      for (int r = 0; r < 8; ++r) {
        float v = mo[r] * 0.8f + co[r] + biaso;
        mo[r] = v;
        outLog[((size_t)(b0 + rowb + r) * T_ + t) * NB_ + w * 16 + nl] = v;
      }
    }
  }

  // ---------------- final states ----------------
#pragma unroll
  for (int r = 0; r < 8; ++r) {
    int row = b0 + rowb + r;
    outM1[(size_t)row * H1_ + w * 32 + nl]      = m1a[r];
    outM1[(size_t)row * H1_ + w * 32 + 16 + nl] = m1b[r];
    outM2[(size_t)row * H2_ + w * 16 + nl]      = m2[r];
  }
  if (w < 4) {
#pragma unroll
    for (int r = 0; r < 8; ++r)
      outMo[(size_t)(b0 + rowb + r) * NB_ + w * 16 + nl] = mo[r];
  }
}

extern "C" void kernel_launch(void* const* d_in, const int* in_sizes, int n_in,
                              void* d_out, int out_size, void* d_ws, size_t ws_size,
                              hipStream_t stream) {
  (void)in_sizes; (void)n_in; (void)out_size;
  const float* x    = (const float*)d_in[0];
  const float* W1   = (const float*)d_in[1];
  const float* b1   = (const float*)d_in[2];
  const float* g1   = (const float*)d_in[3];
  const float* bt1  = (const float*)d_in[4];
  const float* W2   = (const float*)d_in[5];
  const float* b2   = (const float*)d_in[6];
  const float* g2   = (const float*)d_in[7];
  const float* bt2  = (const float*)d_in[8];
  const float* Wout = (const float*)d_in[9];
  const float* bout = (const float*)d_in[10];
  float* out = (float*)d_out;

  const size_t xh_bytes = (size_t)B_ * T_ * F_ * sizeof(_Float16);   // 64 MiB
  const int use_async = (d_ws != nullptr && ws_size >= xh_bytes) ? 1 : 0;
  _Float16* xh = (_Float16*)d_ws;

  if (use_async) {
    // pre-convert x to f16 (one streaming pass; ~8us at 23.3 TB/s)
    const int n8 = B_ * T_ * F_ / 8;                 // elements / 8 per thread
    hipLaunchKernelGGL(cvt_x_kernel, dim3(n8 / 256), dim3(256), 0, stream, x, xh);
  }

  hipLaunchKernelGGL(snn_kernel, dim3(B_ / 16), dim3(256), 0, stream,
                     x, W1, b1, g1, bt1, W2, b2, g2, bt2, Wout, bout,
                     xh, use_async, out);
}
